// HomeostaticSTDP_68994354643157
// MI455X (gfx1250) — matrix operand order repaired and checked by reading, not compiled, not run on hardware
//
#include <hip/hip_runtime.h>

// ---------------------------------------------------------------------------
// HomeostaticSTDP on MI455X (gfx1250).
//   ltp - ltd == (A+ - A-) * sum_{b,t} E[b,t,:] (outer) post[b,t,:]
//   E[t,p] = sum_{tau=1..59} d^tau * pre[t-tau,p],  d = exp(-dt/20)
// The truncated-exponential filter is computed as a banded matmul
//   E[p,t] = sum_i preT[p,i] * M[i,t],  M[i,t] = d^(t+60-i) on the band,
// whose coefficients depend only on (t - i): one 16x96 relative matrix in
// LDS serves every 16-column tile via a shifted window into staged preT.
// Everything (filter + outer-product accumulation) runs on f16 WMMA with
// f32 accumulation; spikes are 0/1 (exact in f16), E is O(1..20) with the
// (A+ - A-) amplitude factored into the scalar epilogue.
// ---------------------------------------------------------------------------

typedef _Float16 v16h __attribute__((ext_vector_type(16)));
typedef _Float16 v8h  __attribute__((ext_vector_type(8)));
typedef float    v8f  __attribute__((ext_vector_type(8)));

constexpr int kB    = 16;
constexpr int kT    = 8192;
constexpr int kN    = 128;            // N_PRE == N_POST
constexpr int kTC   = 64;             // timesteps per LDS chunk (2 WMMA k-steps)
constexpr int kHALO = 60;             // filter history depth
constexpr int kRNG  = 8;              // time ranges per batch
constexpr int kTR   = kT / kRNG;      // 1024 timesteps per block
constexpr int kNBLK = kB * kRNG;      // 128 blocks
constexpr int kSPRE = kTC + kHALO;    // 124 valid pre columns per chunk

constexpr int kPSTR = 152;            // s_preT row stride (halves; 304B, 16B-aligned)
constexpr int kESTR = 72;             // s_Et / s_postT row stride (144B, 16B-aligned)
constexpr int kMSTR = 104;            // s_M row stride (208B, 16B-aligned)

constexpr float A_PLUS  = 0.005f;
constexpr float A_MINUS = 0.00525f;
constexpr float TARGET  = 0.1f;
constexpr float HRATE   = 0.001f;
constexpr float INV_BT  = 1.0f / (float)(kB * kT);

__global__ __launch_bounds__(256)
void stdp_main_kernel(const float* __restrict__ pre,
                      const float* __restrict__ post,
                      const int*   __restrict__ dtp,
                      float* __restrict__ Cpart,      // [kNBLK][128][128]
                      float* __restrict__ postsum2)   // [kNBLK][256]
{
    extern __shared__ _Float16 smem[];
    _Float16* s_preT  = smem;                       // [128][kPSTR] pre^T, chunk window
    _Float16* s_postT = s_preT + kN * kPSTR;        // [128][kESTR] post^T, chunk
    _Float16* s_Et    = s_postT + kN * kESTR;       // [128][kESTR] filtered pre^T
    _Float16* s_M     = s_Et + kN * kESTR;          // [16][kMSTR]  relative filter coeffs

    const int blk  = blockIdx.x;
    const int b    = blk / kRNG;
    const int rng  = blk % kRNG;
    const int t0   = rng * kTR;
    const int tid  = threadIdx.x;
    const int wave = tid >> 5;
    const int lane = tid & 31;
    const int lm   = lane & 15;       // tile row/col within 16
    const int lg   = lane >> 4;       // lane half-group

    const float dt = (float)dtp[0];

    const float* preB  = pre  + (size_t)b * kT * kN;
    const float* postB = post + (size_t)b * kT * kN;

    // ---- relative filter matrix, built once per block -----------------------
    // column tile tc covers t = 16*tc + n; using the i-window [16*tc, 16*tc+95],
    // B[k][n] = s_M[n][k] with k = local i offset, tau = n + 60 - local.
    for (int idx = tid; idx < 16 * kMSTR; idx += 256) {
        int n = idx / kMSTR, local = idx % kMSTR;
        int tau = n + 60 - local;
        float v = (tau >= 1 && tau <= 59) ? __expf(-(float)tau * dt * 0.05f) : 0.0f;
        s_M[idx] = (_Float16)v;
    }

    float psum = 0.0f;                // this thread owns post channel (tid & 127)

    v8f acc[8];
    const v8f zf = {0.f, 0.f, 0.f, 0.f, 0.f, 0.f, 0.f, 0.f};
#pragma unroll
    for (int c = 0; c < 8; ++c) acc[c] = zf;

    const int nchunks = kTR / kTC;
    for (int chunk = 0; chunk < nchunks; ++chunk) {
        const int c0 = t0 + chunk * kTC;
        __syncthreads();              // protect LDS reuse from previous readers

        // prefetch next chunk's spike data (global_prefetch_b8)
        if (chunk + 1 < nchunks) {
            __builtin_prefetch(preB  + (size_t)(c0 + kTC - kHALO) * kN + (size_t)tid * 62, 0, 0);
            __builtin_prefetch(postB + (size_t)(c0 + kTC) * kN + (size_t)tid * 32, 0, 0);
        }

        // stage pre^T: columns i in [0,124) <- global t = c0-60+i; zero-pad to kPSTR
        for (int idx = tid; idx < kPSTR * kN; idx += 256) {
            int i = idx >> 7, p = idx & 127;          // consecutive tids -> consecutive p
            int gt = c0 - kHALO + i;
            float v = (i < kSPRE && gt >= 0) ? preB[(size_t)gt * kN + p] : 0.0f;
            s_preT[p * kPSTR + i] = (_Float16)v;
        }
        // stage post^T + homeostatic rate partial (each thread owns q = tid&127)
        for (int idx = tid; idx < kTC * kN; idx += 256) {
            int trow = idx >> 7, q = idx & 127;
            float v = postB[(size_t)(c0 + trow) * kN + q];
            psum += v;
            s_postT[q * kESTR + trow] = (_Float16)v;
        }
        __syncthreads();

        // ---- filter GEMM: E[p, t] = preT x M, banded window per column tile --
        const _Float16* aPbase = s_preT + (wave * 16 + lm) * kPSTR;
#pragma unroll
        for (int tc = 0; tc < 4; ++tc) {
            v16h ea[3], eb[3];
#pragma unroll
            for (int ks = 0; ks < 3; ++ks) {
                const _Float16* ap = aPbase + tc * 16 + ks * 32;
                ((v8h*)&ea[ks])[0] = *(const v8h*)(ap + lg * 8);
                ((v8h*)&ea[ks])[1] = *(const v8h*)(ap + 16 + lg * 8);
                const _Float16* bp = s_M + lm * kMSTR + ks * 32;
                ((v8h*)&eb[ks])[0] = *(const v8h*)(bp + lg * 8);
                ((v8h*)&eb[ks])[1] = *(const v8h*)(bp + 16 + lg * 8);
            }
            v8f accE = zf;
#pragma unroll
            for (int ks = 0; ks < 3; ++ks)
                accE = __builtin_amdgcn_wmma_f32_16x16x32_f16(
                    false, ea[ks], false, eb[ks], (short)0, accE, false, false);
            // store E tile: this wave's own rows only -> same-wave DS ordering
            // (LDS ops are in-order within a wave), no extra barrier needed.
            _Float16* ep = s_Et + (wave * 16 + lg * 8) * kESTR + tc * 16 + lm;
#pragma unroll
            for (int rr = 0; rr < 8; ++rr) ep[rr * kESTR] = (_Float16)accE[rr];
        }

        // ---- outer-product GEMM: acc += E^T-tile x post^T-tile ---------------
        const _Float16* arow = s_Et + (wave * 16 + lm) * kESTR;
#pragma unroll
        for (int ks = 0; ks < 2; ++ks) {
            const int kbase = ks * 32;
            v16h a;
            ((v8h*)&a)[0] = *(const v8h*)(arow + kbase + lg * 8);
            ((v8h*)&a)[1] = *(const v8h*)(arow + kbase + 16 + lg * 8);
            v16h bf[8];
#pragma unroll
            for (int c = 0; c < 8; ++c) {   // preload all B frags: overlap DS w/ WMMA
                const _Float16* brow = s_postT + (c * 16 + lm) * kESTR + kbase;
                ((v8h*)&bf[c])[0] = *(const v8h*)(brow + lg * 8);
                ((v8h*)&bf[c])[1] = *(const v8h*)(brow + 16 + lg * 8);
            }
#pragma unroll
            for (int c = 0; c < 8; ++c)
                acc[c] = __builtin_amdgcn_wmma_f32_16x16x32_f16(
                    false, a, false, bf[c], (short)0, acc[c], false, false);
        }
    }

    // per-block partial C: C/D layout -> element (M = rr + 8*lg, N = lm) in VGPR rr
    float* cp = Cpart + (size_t)blk * kN * kN;
#pragma unroll
    for (int c = 0; c < 8; ++c) {
        float* dst = cp + (wave * 16 + lg * 8) * kN + c * 16 + lm;
#pragma unroll
        for (int rr = 0; rr < 8; ++rr) dst[rr * kN] = acc[c][rr];
    }
    postsum2[blk * 256 + tid] = psum;
}

__global__ __launch_bounds__(256)
void stdp_reduce_kernel(const float* __restrict__ Cpart,
                        const float* __restrict__ postsum2,
                        const float* __restrict__ W,
                        const int*   __restrict__ dtp,
                        float* __restrict__ out)
{
    const int idx = blockIdx.x * 256 + threadIdx.x;   // 0..16383
    const int q = idx & 127;

    float s = 0.0f;
    for (int i = 0; i < kNBLK; ++i) s += Cpart[(size_t)i * kN * kN + idx];
    float ps = 0.0f;
    for (int i = 0; i < kNBLK; ++i)
        ps += postsum2[i * 256 + q] + postsum2[i * 256 + 128 + q];

    const float dt    = (float)dtp[0];
    const float alpha = dt * 0.001f;                  // dt / TAU_HOMEOSTATIC
    const float stdpv = (A_PLUS - A_MINUS) * s * INV_BT;
    const float homeo = -HRATE * alpha * (ps * INV_BT - TARGET) * W[idx];
    out[idx] = stdpv + homeo;
}

extern "C" void kernel_launch(void* const* d_in, const int* in_sizes, int n_in,
                              void* d_out, int out_size, void* d_ws, size_t ws_size,
                              hipStream_t stream) {
    const float* pre  = (const float*)d_in[0];
    const float* post = (const float*)d_in[1];
    const float* W    = (const float*)d_in[2];
    const int*   dtp  = (const int*)d_in[3];
    float* out = (float*)d_out;

    // workspace: [128][128][128] f32 partials (8 MB) + [128][256] rate partials
    float* Cpart    = (float*)d_ws;
    float* postsum2 = Cpart + (size_t)kNBLK * kN * kN;

    const size_t lds_bytes =
        (size_t)(kN * kPSTR + 2 * kN * kESTR + 16 * kMSTR) * sizeof(_Float16); // ~77 KB (<320 KB WGP)

    stdp_main_kernel<<<dim3(kNBLK), dim3(256), lds_bytes, stream>>>(
        pre, post, dtp, Cpart, postsum2);
    stdp_reduce_kernel<<<dim3((kN * kN) / 256), dim3(256), 0, stream>>>(
        Cpart, postsum2, W, dtp, out);
}
